// MVOnlyGATrAttention_82678120448616
// MI455X (gfx1250) — matrix operand order repaired and checked by reading, not compiled, not run on hardware
//
#include <hip/hip_runtime.h>
#include <hip/hip_bf16.h>

#define BB 4
#define TT 2048
#define CC 32
#define HH 4
#define FQK 416        // 13 * 32 attention feature dims (ipa 256 + daa 160)
#define FV  512        // C*16 value dims
#define QF_STRIDE 424  // padded LDS row stride (conflict-free across 16 rows)

typedef __bf16 bf16;
typedef __attribute__((ext_vector_type(16))) __bf16 v16bf;
typedef __attribute__((ext_vector_type(8)))  __bf16 v8bf;
typedef __attribute__((ext_vector_type(8)))  float  v8f;

__constant__ int d_GRADE[16] = {0,1,1,1,1,2,2,2,2,2,2,3,3,3,3,4};
__constant__ int d_E0SRC[16] = {-1,0,-1,-1,-1,2,3,4,-1,-1,-1,8,9,10,-1,14};
__constant__ int d_E0B[16]   = {-1,5,-1,-1,-1,6,6,6,-1,-1,-1,7,7,7,-1,8};
__constant__ int d_INV[8]    = {0,2,3,4,8,9,10,14};

// ---- WMMA fragment gathers (layouts per CDNA5 ISA 7.12.2) ----
// All call sites guarantee 16-byte alignment -> b128 loads.

// B matrix 32x16 (KxN), 16-bit: lane<16 -> N=lane, K {0..7,16..23}; lane>=16 -> K {8..15,24..31}.
__device__ inline v16bf fragB16(const bf16* rowptr, int khalf) {
  v8bf lo = *(const v8bf*)(rowptr + khalf * 8);
  v8bf hi = *(const v8bf*)(rowptr + 16 + khalf * 8);
  return __builtin_shufflevector(lo, hi, 0, 1, 2, 3, 4, 5, 6, 7, 8, 9, 10, 11,
                                 12, 13, 14, 15);
}

// A matrix 16x32 (MxK), 16-bit: lane<16 -> M=lane, K {0..7,16..23}; lane>=16 -> K {8..15,24..31}.
__device__ inline v16bf fragA(const bf16* rowptr, int khalf) {
  v8bf lo = *(const v8bf*)(rowptr + khalf * 8);
  v8bf hi = *(const v8bf*)(rowptr + 16 + khalf * 8);
  return __builtin_shufflevector(lo, hi, 0, 1, 2, 3, 4, 5, 6, 7, 8, 9, 10, 11,
                                 12, 13, 14, 15);
}

__device__ inline v8f wmma_bf16(v16bf a, v16bf b, v8f c) {
  return __builtin_amdgcn_wmma_f32_16x16x32_bf16(false, a, false, b, (short)0, c,
                                                 false, false);
}

// =====================================================================
// Kernel 0: pre-pack equi-linear weights into WMMA-friendly bf16 layout
//   wqkvP[bsel][o][k] (o<384,k<32), woutP[bsel][o][k] (o<32,k<128)
// =====================================================================
__global__ __launch_bounds__(256) void k_pack_w(const float* __restrict__ wqkv,
                                                const float* __restrict__ wout,
                                                bf16* __restrict__ wqkvP,
                                                bf16* __restrict__ woutP) {
  int tid = blockIdx.x * 256 + threadIdx.x;
  int stride = gridDim.x * 256;
  for (int e = tid; e < 9 * 384 * 32; e += stride) {
    int bsel = e / (384 * 32);
    int r = e % (384 * 32);
    int o = r >> 5, k = r & 31;
    wqkvP[e] = (bf16)wqkv[((size_t)o * 32 + k) * 9 + bsel];
  }
  for (int e = tid; e < 9 * 32 * 128; e += stride) {
    int bsel = e / (32 * 128);
    int r = e % (32 * 128);
    int o = r >> 7, k = r & 127;
    woutP[e] = (bf16)wout[((size_t)o * 128 + k) * 9 + bsel];
  }
}

// =====================================================================
// Kernel 1: RMS norm + equi_linear(qkv) via WMMA + feature construction
// =====================================================================
__global__ __launch_bounds__(128) void k_qkv_features(
    const float* __restrict__ x, const float* __restrict__ lnw,
    const bf16* __restrict__ wqkvP, const float* __restrict__ mix_ipa,
    const float* __restrict__ mix_daa, bf16* __restrict__ qf,
    bf16* __restrict__ kf, bf16* __restrict__ vfT) {
  __shared__ __align__(16) float xt[16 * 512];      // [tok][c*16+j]
  __shared__ __align__(16) bf16 xn2[16 * 16 * 32];  // [blade][tok][c]
  __shared__ __align__(16) float qbuf[16 * 32 * 16];// [tok][ch][blade]
  __shared__ float rr[16];

  int blk = blockIdx.x;
  int b = blk / (TT / 16);
  int t0 = (blk % (TT / 16)) * 16;
  int tid = threadIdx.x;
  int lane = tid & 31;
  int wv = tid >> 5;
  int khalf = lane >> 4;
  int ln15 = lane & 15;

  for (int e = tid; e < 16 * 512; e += 128) {
    int m = e >> 9, r = e & 511;
    xt[e] = x[((size_t)(b * TT + t0 + m)) * 512 + r];
  }
  __syncthreads();
  if (tid < 16) {
    float s = 0.f;
    for (int c = 0; c < 32; ++c) {
      const float* p = &xt[tid * 512 + c * 16];
#pragma unroll
      for (int ii = 0; ii < 8; ++ii) {
        float v = p[d_INV[ii]];
        s += v * v;
      }
    }
    rr[tid] = rsqrtf(s * (1.0f / 32.0f) + 1e-5f);
  }
  __syncthreads();
  for (int e = tid; e < 16 * 512; e += 128) {
    int m = e >> 9, r = e & 511;
    int c = r >> 4, j = r & 15;
    xn2[(j * 16 + m) * 32 + c] = (bf16)(xt[e] * rr[m] * lnw[c]);
  }
  __syncthreads();

  for (int g = 0; g < 12; ++g) {    // 3 (q,k,v) x 4 heads
    int s3 = g >> 2;                // 0=q 1=k 2=v
    int h = g & 3;
    int obase = s3 * 128 + h * 32;  // output-channel base in [384]
    for (int job = wv; job < 32; job += 4) {  // (blade j, ntile nt)
      int j = job >> 1;
      int nt = job & 1;
      v8f acc = {};
      v16bf a = fragA(&xn2[(j * 16 + ln15) * 32], khalf);
      v16bf wb = fragB16(
          wqkvP + ((size_t)d_GRADE[j] * 384 + obase + nt * 16 + ln15) * 32, khalf);
      acc = wmma_bf16(a, wb, acc);
      int src = d_E0SRC[j];
      if (src >= 0) {
        v16bf a2 = fragA(&xn2[(src * 16 + ln15) * 32], khalf);
        v16bf wb2 = fragB16(
            wqkvP + ((size_t)d_E0B[j] * 384 + obase + nt * 16 + ln15) * 32, khalf);
        acc = wmma_bf16(a2, wb2, acc);
      }
#pragma unroll
      for (int i = 0; i < 8; ++i) {
        int m = i + 8 * khalf;
        int ch = nt * 16 + ln15;
        qbuf[(m * 32 + ch) * 16 + j] = acc[i];
      }
    }
    __syncthreads();

    size_t bh = (size_t)(b * HH + h);
    if (s3 == 0) {  // q features (with exp(mix) weights)
      for (int e = tid; e < 16 * FQK; e += 128) {
        int m = e / FQK, f = e % FQK;
        float val;
        if (f < 256) {
          int c = f >> 3, ii = f & 7;
          val = qbuf[(m * 32 + c) * 16 + d_INV[ii]] * __expf(mix_ipa[h * CC + c]);
        } else {
          int f2 = f - 256;
          int c = f2 / 5, d = f2 % 5;
          float wd = __expf(mix_daa[h * CC + c]);
          float p0 = qbuf[(m * 32 + c) * 16 + 11];
          float p1 = qbuf[(m * 32 + c) * 16 + 12];
          float p2 = qbuf[(m * 32 + c) * 16 + 13];
          if (d < 3)
            val = 2.0f * ((d == 0) ? p0 : (d == 1) ? p1 : p2) * wd;
          else if (d == 3)
            val = -(p0 * p0 + p1 * p1 + p2 * p2) * wd;
          else
            val = -wd;
        }
        qf[(bh * TT + t0 + m) * FQK + f] = (bf16)val;
      }
    } else if (s3 == 1) {  // k features
      for (int e = tid; e < 16 * FQK; e += 128) {
        int m = e / FQK, f = e % FQK;
        float val;
        if (f < 256) {
          int c = f >> 3, ii = f & 7;
          val = qbuf[(m * 32 + c) * 16 + d_INV[ii]];
        } else {
          int f2 = f - 256;
          int c = f2 / 5, d = f2 % 5;
          float p0 = qbuf[(m * 32 + c) * 16 + 11];
          float p1 = qbuf[(m * 32 + c) * 16 + 12];
          float p2 = qbuf[(m * 32 + c) * 16 + 13];
          if (d < 3)
            val = (d == 0) ? p0 : (d == 1) ? p1 : p2;
          else if (d == 3)
            val = 1.0f;
          else
            val = p0 * p0 + p1 * p1 + p2 * p2;
        }
        kf[(bh * TT + t0 + m) * FQK + f] = (bf16)val;
      }
    } else {  // v features, stored transposed [bh][512][T]
      for (int e = tid; e < 16 * FV; e += 128) {
        int m = e >> 9, f = e & 511;
        int c = f >> 4, j = f & 15;
        vfT[(bh * FV + f) * TT + t0 + m] = (bf16)qbuf[(m * 32 + c) * 16 + j];
      }
    }
    __syncthreads();
  }
}

// =====================================================================
// Kernel 2: causal flash attention, bf16 WMMA, 64 queries / block
// =====================================================================
__global__ __launch_bounds__(256) void k_attention(
    const bf16* __restrict__ qf, const bf16* __restrict__ kf,
    const bf16* __restrict__ vfT, bf16* __restrict__ OblT) {
  __shared__ __align__(16) bf16 qfs[64 * QF_STRIDE];
  __shared__ __align__(16) bf16 Pbuf[4 * 16 * 32];
  __shared__ float mrow[4][16], lrow[4][16], alph[4][16];
  __shared__ float pmax[4][2][16], psum[4][2][16];

  int blk = blockIdx.x;
  int nq = TT / 64;
  int bh = blk / nq;
  int qb = (blk % nq) * 64;
  int tid = threadIdx.x, lane = tid & 31, wv = tid >> 5;
  int mt = wv >> 1, sub = wv & 1;
  int khalf = lane >> 4, ln15 = lane & 15;

  const bf16* qg = qf + ((size_t)bh * TT + qb) * FQK;
  for (int e = tid; e < 64 * FQK; e += 256) {
    int r = e / FQK, f = e % FQK;
    qfs[r * QF_STRIDE + f] = qg[(size_t)r * FQK + f];
  }
  if (tid < 64) {
    mrow[tid >> 4][tid & 15] = -1e30f;
    lrow[tid >> 4][tid & 15] = 0.f;
  }
  __syncthreads();

  v8f accO[16];
#pragma unroll
  for (int i = 0; i < 16; ++i) accO[i] = (v8f){};

  const float scale = 0.049029034f;  // 1/sqrt(416)
  int kend = qb + 64;
  for (int kb = 0; kb < kend; kb += 32) {
    if (kb + 32 < kend)
      __builtin_prefetch(kf + ((size_t)bh * TT + kb + 32 + sub * 16 + ln15) * FQK,
                         0, 0);
    // ---- scores: Qf(16x416) x Kf^T(416x16) for this wave's 16 keys ----
    v8f S = {};
    const bf16* krow = kf + ((size_t)bh * TT + kb + sub * 16 + ln15) * FQK;
#pragma unroll
    for (int ks = 0; ks < 13; ++ks) {
      v16bf a = fragA(&qfs[(mt * 16 + ln15) * QF_STRIDE + ks * 32], khalf);
      v16bf bm = fragB16(krow + ks * 32, khalf);
      S = wmma_bf16(a, bm, S);
    }
    // scale + causal mask
    float Sv[8];
#pragma unroll
    for (int i = 0; i < 8; ++i) {
      int m = i + 8 * khalf;
      int qidx = qb + mt * 16 + m;
      int kidx = kb + sub * 16 + ln15;
      float s = S[i] * scale;
      Sv[i] = (kidx > qidx) ? -1e30f : s;
    }
    // per-wave row max (rows live across 16-lane groups)
    float rmx[8];
#pragma unroll
    for (int i = 0; i < 8; ++i) {
      float v = Sv[i];
      v = fmaxf(v, __shfl_xor(v, 1));
      v = fmaxf(v, __shfl_xor(v, 2));
      v = fmaxf(v, __shfl_xor(v, 4));
      v = fmaxf(v, __shfl_xor(v, 8));
      rmx[i] = v;
    }
    if (ln15 == 0) {
#pragma unroll
      for (int i = 0; i < 8; ++i) pmax[mt][sub][i + 8 * khalf] = rmx[i];
    }
    __syncthreads();
    if (sub == 0 && lane < 16) {
      float mo = mrow[mt][lane];
      float mn = fmaxf(mo, fmaxf(pmax[mt][0][lane], pmax[mt][1][lane]));
      mrow[mt][lane] = mn;
      alph[mt][lane] = __expf(mo - mn);
    }
    __syncthreads();
    // P = exp(S - m), row sums, write P tile, rescale O accumulators
    float rs[8];
#pragma unroll
    for (int i = 0; i < 8; ++i) {
      int m = i + 8 * khalf;
      float p = __expf(Sv[i] - mrow[mt][m]);
      Pbuf[(mt * 16 + m) * 32 + sub * 16 + ln15] = (bf16)p;
      float v = p;
      v += __shfl_xor(v, 1);
      v += __shfl_xor(v, 2);
      v += __shfl_xor(v, 4);
      v += __shfl_xor(v, 8);
      rs[i] = v;
    }
    if (ln15 == 0) {
#pragma unroll
      for (int i = 0; i < 8; ++i) psum[mt][sub][i + 8 * khalf] = rs[i];
    }
    float al[8];
#pragma unroll
    for (int i = 0; i < 8; ++i) al[i] = alph[mt][i + 8 * khalf];
#pragma unroll
    for (int nt = 0; nt < 16; ++nt)
#pragma unroll
      for (int i = 0; i < 8; ++i) accO[nt][i] *= al[i];
    __syncthreads();
    if (sub == 0 && lane < 16) {
      lrow[mt][lane] =
          lrow[mt][lane] * alph[mt][lane] + psum[mt][0][lane] + psum[mt][1][lane];
    }
    // ---- O += P(16x32) x V(32x256-per-wave) ----
#pragma unroll
    for (int nt = 0; nt < 16; ++nt) {
      int fbase = sub * 256 + nt * 16;
      v16bf a = fragA(&Pbuf[(mt * 16 + ln15) * 32], khalf);
      const bf16* vrow = vfT + ((size_t)bh * FV + fbase + ln15) * TT + kb;
      v16bf bm = fragB16(vrow, khalf);
      accO[nt] = wmma_bf16(a, bm, accO[nt]);
    }
    __syncthreads();
  }
  __syncthreads();
  float li[8];
#pragma unroll
  for (int i = 0; i < 8; ++i) li[i] = 1.0f / lrow[mt][i + 8 * khalf];
  // store blade-major: OblT[bh][blade j][t][c]
#pragma unroll
  for (int nt = 0; nt < 16; ++nt) {
    int c = sub * 16 + nt;  // f = sub*256 + nt*16 + n -> c = f>>4, j = f&15 = n
    int j = ln15;
#pragma unroll
    for (int i = 0; i < 8; ++i) {
      int m = i + 8 * khalf;
      int t = qb + mt * 16 + m;
      OblT[(((size_t)bh * 16 + j) * TT + t) * 32 + c] = (bf16)(accO[nt][i] * li[i]);
    }
  }
}

// =====================================================================
// Kernel 3: output equi_linear (K=128) via WMMA + residual
// =====================================================================
__global__ __launch_bounds__(128) void k_outproj(const bf16* __restrict__ OblT,
                                                 const bf16* __restrict__ woutP,
                                                 const float* __restrict__ x,
                                                 float* __restrict__ out) {
  int blk = blockIdx.x;
  int b = blk / (TT / 16);
  int t0 = (blk % (TT / 16)) * 16;
  int tid = threadIdx.x, lane = tid & 31, wv = tid >> 5;
  int khalf = lane >> 4, ln15 = lane & 15;
  for (int job = wv; job < 32; job += 4) {
    int j = job >> 1, nt = job & 1;
    v8f acc = {};
#pragma unroll
    for (int kc = 0; kc < 4; ++kc) {  // K = 128 in-channels = 4 heads x 32
      const bf16* arow =
          OblT + (((size_t)(b * HH + kc) * 16 + j) * TT + t0 + ln15) * 32;
      v16bf a = fragA(arow, khalf);
      v16bf wb = fragB16(
          woutP + ((size_t)d_GRADE[j] * 32 + nt * 16 + ln15) * 128 + kc * 32, khalf);
      acc = wmma_bf16(a, wb, acc);
    }
    int src = d_E0SRC[j];
    if (src >= 0) {
#pragma unroll
      for (int kc = 0; kc < 4; ++kc) {
        const bf16* arow =
            OblT + (((size_t)(b * HH + kc) * 16 + src) * TT + t0 + ln15) * 32;
        v16bf a = fragA(arow, khalf);
        v16bf wb = fragB16(
            woutP + ((size_t)d_E0B[j] * 32 + nt * 16 + ln15) * 128 + kc * 32, khalf);
        acc = wmma_bf16(a, wb, acc);
      }
    }
#pragma unroll
    for (int i = 0; i < 8; ++i) {
      int m = i + 8 * khalf;
      int ch = nt * 16 + ln15;
      size_t idx = ((size_t)(b * TT + t0 + m)) * 512 + ch * 16 + j;
      out[idx] = acc[i] + x[idx];
    }
  }
}

extern "C" void kernel_launch(void* const* d_in, const int* in_sizes, int n_in,
                              void* d_out, int out_size, void* d_ws,
                              size_t ws_size, hipStream_t stream) {
  const float* x = (const float*)d_in[0];
  const float* lnw = (const float*)d_in[1];
  const float* wqkv = (const float*)d_in[2];
  const float* wout = (const float*)d_in[3];
  const float* mix_ipa = (const float*)d_in[4];
  const float* mix_daa = (const float*)d_in[5];
  float* out = (float*)d_out;

  char* ws = (char*)d_ws;
  size_t SQ = (size_t)BB * HH * TT * FQK * sizeof(bf16);  // qf / kf
  size_t SV = (size_t)BB * HH * TT * FV * sizeof(bf16);   // vfT / OblT
  bf16* qf = (bf16*)ws;
  bf16* kf = (bf16*)(ws + SQ);
  bf16* vfT = (bf16*)(ws + 2 * SQ);
  bf16* OblT = (bf16*)(ws + 2 * SQ + SV);
  bf16* wqkvP = (bf16*)(ws + 2 * SQ + 2 * SV);
  bf16* woutP = wqkvP + (size_t)9 * 384 * 32;

  k_pack_w<<<64, 256, 0, stream>>>(wqkv, wout, wqkvP, woutP);
  k_qkv_features<<<BB * (TT / 16), 128, 0, stream>>>(x, lnw, wqkvP, mix_ipa,
                                                     mix_daa, qf, kf, vfT);
  k_attention<<<BB * HH * (TT / 64), 256, 0, stream>>>(qf, kf, vfT, OblT);
  k_outproj<<<BB * (TT / 16), 128, 0, stream>>>(OblT, woutP, x, out);
}
